// HypergraphConv_62792421867972
// MI455X (gfx1250) — matrix operand order repaired and checked by reading, not compiled, not run on hardware
//
#include <hip/hip_runtime.h>
#include <hip/hip_bf16.h>

typedef __attribute__((ext_vector_type(16))) _Float16 v16h;
typedef __attribute__((ext_vector_type(8)))  float    v8f;
typedef __attribute__((ext_vector_type(4)))  _Float16 h4;

#define B_DIM 2
#define C_DIM 128
#define N_PTS 50000
#define K_HE  5000
#define K_PAD 5008      // 16 * 313
#define M1    32
#define M2    8
#define LDA   136       // LDS tile row stride in halves: 272B, 16B-aligned,
                        // rows start at distinct banks (4*m mod 64) -> no b128 conflicts

// ---------------------------------------------------------------------------
// Weight convert: w [In][Out] f32 row-major  ->  wT [Out][In] f16 row-major
// ---------------------------------------------------------------------------
__global__ void convert_wT_kernel(const float* __restrict__ w, _Float16* __restrict__ wT) {
    int idx = blockIdx.x * 256 + threadIdx.x;              // over 128*128
    if (idx >= C_DIM * C_DIM) return;
    int o = idx >> 7, i = idx & 127;                       // wT[o][i]
    wT[idx] = (_Float16)w[i * C_DIM + o];
}

// ---------------------------------------------------------------------------
// Tiled transposes  in [B][128][R] f32  ->  out [B][Rpad][128]
// rows r in [R, Rpad) zero-filled.
// ---------------------------------------------------------------------------
__global__ void transpose_f16_kernel(const float* __restrict__ in, _Float16* __restrict__ out,
                                     int R, int Rpad) {
    __shared__ float tile[32][33];
    int b = blockIdx.z;
    int r0 = blockIdx.x * 32, c0 = blockIdx.y * 32;
    int tx = threadIdx.x, ty = threadIdx.y;                // block (32, 8)
    for (int i = ty; i < 32; i += 8) {
        int c = c0 + i, r = r0 + tx;
        tile[i][tx] = (r < R) ? in[((size_t)b * C_DIM + c) * R + r] : 0.0f;
    }
    __syncthreads();
    for (int i = ty; i < 32; i += 8) {
        int r = r0 + i, c = c0 + tx;
        if (r < Rpad) out[((size_t)b * Rpad + r) * C_DIM + c] = (_Float16)tile[tx][i];
    }
}

__global__ void transpose_f32_kernel(const float* __restrict__ in, float* __restrict__ out,
                                     int R, int Rpad) {
    __shared__ float tile[32][33];
    int b = blockIdx.z;
    int r0 = blockIdx.x * 32, c0 = blockIdx.y * 32;
    int tx = threadIdx.x, ty = threadIdx.y;
    for (int i = ty; i < 32; i += 8) {
        int c = c0 + i, r = r0 + tx;
        tile[i][tx] = (r < R) ? in[((size_t)b * C_DIM + c) * R + r] : 0.0f;
    }
    __syncthreads();
    for (int i = ty; i < 32; i += 8) {
        int r = r0 + i, c = c0 + tx;
        if (r < Rpad) out[((size_t)b * Rpad + r) * C_DIM + c] = tile[tx][i];
    }
}

// ---------------------------------------------------------------------------
// WMMA fragment builders (CDNA5 16x16x32 f16 layouts, wave32)
// ---------------------------------------------------------------------------
// A (16x32): lane L -> row M = L&15; VGPR v pair K = kb*32 + 8*(L>>4) + {2v | 16+2(v-4)}
// T is an LDS tile with row stride LDA halves; each 4-dword group is contiguous
// -> lowers to two bank-conflict-free ds_load_b128 per fragment.
__device__ __forceinline__ v16h a_frag(const _Float16* T, int lane, int kb) {
    int m = lane & 15, g = lane >> 4;
    int base = kb * 32 + 8 * g;
    union { v16h v; unsigned u[8]; } f;
    const unsigned* row = (const unsigned*)(T + m * LDA);
    #pragma unroll
    for (int r = 0; r < 8; ++r) {
        int kk = base + 2 * (r & 3) + ((r >> 2) << 4);
        f.u[r] = row[kk >> 1];
    }
    return f.v;
}

// B (32x16): lane L -> col N = n0 + (L&15); VGPR v pair K = kb*32 + 8*(L>>4) + {2v | 16+2(v-4)}
// wT is [col][128] f16 row-major -> two runs of 4 contiguous dwords (fuses to global b128 pairs).
__device__ __forceinline__ v16h b_frag(const _Float16* __restrict__ wT, int lane, int n0, int kb) {
    int n = n0 + (lane & 15), g = lane >> 4;
    const unsigned* row = (const unsigned*)(wT + n * C_DIM + kb * 32 + g * 8);
    union { v16h v; unsigned u[8]; } f;
    #pragma unroll
    for (int r = 0; r < 8; ++r) f.u[r] = row[(r & 3) + ((r >> 2) << 3)];
    return f.v;
}

// Gather one 256B source row set into a float4 accumulator (4 channels per lane).
__device__ __forceinline__ void gather_row(const _Float16* __restrict__ src, size_t row_base,
                                           int myidx, int nidx,
                                           float& a0, float& a1, float& a2, float& a3) {
    for (int m = 0; m < nidx; ++m) {
        int id = __shfl(myidx, m, 32);
        if (id >= 0) {
            h4 v = *(const h4*)(src + (row_base + (size_t)id) * C_DIM + (threadIdx.x & 31) * 4);
            a0 += (float)v[0]; a1 += (float)v[1]; a2 += (float)v[2]; a3 += (float)v[3];
        }
    }
}

// ---------------------------------------------------------------------------
// Fused kernel 1: gather(hm) -> relu(A@w1a+b1a)@w1b + b1b + (1+eps)*centersT
// One block = 16-row strip x 128 cols; 8 waves: gather 2 rows each into LDS,
// then one 16x16 WMMA tile each per phase.
// ---------------------------------------------------------------------------
__global__ void gemm_he_fused(const _Float16* __restrict__ nfT, const int* __restrict__ hm,
                              const _Float16* __restrict__ w1aT, const float* __restrict__ b1a,
                              const _Float16* __restrict__ w1bT, const float* __restrict__ b1b,
                              const float* __restrict__ cenT, const float* __restrict__ epsp,
                              _Float16* __restrict__ heh) {
    __shared__ __align__(16) _Float16 At[16 * LDA];
    __shared__ __align__(16) _Float16 Ht[16 * LDA];
    const int strips = K_PAD / 16;
    int b = blockIdx.x / strips, strip = blockIdx.x % strips;
    int k0 = strip * 16;
    int lane = threadIdx.x & 31, wave = threadIdx.x >> 5;

    // ---- gather A tile (fused node->hyperedge sum), f16 into LDS ----
    #pragma unroll
    for (int j = 0; j < 2; ++j) {
        int row = 2 * wave + j;
        int k = k0 + row;
        float a0 = 0.f, a1 = 0.f, a2 = 0.f, a3 = 0.f;
        if (k < K_HE) {
            int myidx = hm[((size_t)b * K_HE + k) * M1 + lane];
            gather_row(nfT, (size_t)b * N_PTS, myidx, M1, a0, a1, a2, a3);
        }
        h4 o; o[0] = (_Float16)a0; o[1] = (_Float16)a1; o[2] = (_Float16)a2; o[3] = (_Float16)a3;
        *(h4*)(At + row * LDA + lane * 4) = o;
    }
    __syncthreads();

    int n0 = wave * 16, n = n0 + (lane & 15), g = lane >> 4;

    v8f acc;
    float bn = b1a[n];
    #pragma unroll
    for (int r = 0; r < 8; ++r) acc[r] = bn;
    #pragma unroll
    for (int kb = 0; kb < 4; ++kb) {
        v16h a  = a_frag(At, lane, kb);
        v16h bm = b_frag(w1aT, lane, n0, kb);
        acc = __builtin_amdgcn_wmma_f32_16x16x32_f16(false, a, false, bm, (short)0, acc, false, false);
    }
    #pragma unroll
    for (int r = 0; r < 8; ++r)                            // relu -> H tile (f16)
        Ht[(r + 8 * g) * LDA + n] = (_Float16)fmaxf(acc[r], 0.0f);
    __syncthreads();

    v8f acc2;
    float bn2 = b1b[n];
    #pragma unroll
    for (int r = 0; r < 8; ++r) acc2[r] = bn2;
    #pragma unroll
    for (int kb = 0; kb < 4; ++kb) {
        v16h a  = a_frag(Ht, lane, kb);
        v16h bm = b_frag(w1bT, lane, n0, kb);
        acc2 = __builtin_amdgcn_wmma_f32_16x16x32_f16(false, a, false, bm, (short)0, acc2, false, false);
    }
    float ep = 1.0f + epsp[0];
    size_t rowbase = ((size_t)b * K_PAD + k0) * C_DIM;
    #pragma unroll
    for (int r = 0; r < 8; ++r) {
        size_t ridx = rowbase + (size_t)(r + 8 * g) * C_DIM + n;
        heh[ridx] = (_Float16)(acc2[r] + ep * cenT[ridx]);
    }
}

// ---------------------------------------------------------------------------
// Fused kernel 2: gather(phi) -> relu(A@w2a+b2a)@w2b + b2b, stored transposed
// into out[b][o][n]. heh (2.6MB) is L2-resident -> gather reads stay on-chip.
// ---------------------------------------------------------------------------
__global__ void gemm_out_fused(const _Float16* __restrict__ heh, const int* __restrict__ phi,
                               const _Float16* __restrict__ w2aT, const float* __restrict__ b2a,
                               const _Float16* __restrict__ w2bT, const float* __restrict__ b2b,
                               float* __restrict__ out) {
    __shared__ __align__(16) _Float16 At[16 * LDA];
    __shared__ __align__(16) _Float16 Ht[16 * LDA];
    const int strips = N_PTS / 16;                         // 3125 exact
    int b = blockIdx.x / strips, strip = blockIdx.x % strips;
    int p0 = strip * 16;
    int lane = threadIdx.x & 31, wave = threadIdx.x >> 5;

    // ---- gather A tile (fused hyperedge->node sum), f16 into LDS ----
    #pragma unroll
    for (int j = 0; j < 2; ++j) {
        int row = 2 * wave + j;
        int p = p0 + row;
        float a0 = 0.f, a1 = 0.f, a2 = 0.f, a3 = 0.f;
        int myidx = (lane < M2) ? phi[((size_t)b * N_PTS + p) * M2 + lane] : -1;
        gather_row(heh, (size_t)b * K_PAD, myidx, M2, a0, a1, a2, a3);
        h4 o; o[0] = (_Float16)a0; o[1] = (_Float16)a1; o[2] = (_Float16)a2; o[3] = (_Float16)a3;
        *(h4*)(At + row * LDA + lane * 4) = o;
    }
    __syncthreads();

    int n0 = wave * 16, n = n0 + (lane & 15), g = lane >> 4;

    v8f acc;
    float bn = b2a[n];
    #pragma unroll
    for (int r = 0; r < 8; ++r) acc[r] = bn;
    #pragma unroll
    for (int kb = 0; kb < 4; ++kb) {
        v16h a  = a_frag(At, lane, kb);
        v16h bm = b_frag(w2aT, lane, n0, kb);
        acc = __builtin_amdgcn_wmma_f32_16x16x32_f16(false, a, false, bm, (short)0, acc, false, false);
    }
    #pragma unroll
    for (int r = 0; r < 8; ++r)
        Ht[(r + 8 * g) * LDA + n] = (_Float16)fmaxf(acc[r], 0.0f);
    __syncthreads();

    v8f acc2;
    float bn2 = b2b[n];
    #pragma unroll
    for (int r = 0; r < 8; ++r) acc2[r] = bn2;
    #pragma unroll
    for (int kb = 0; kb < 4; ++kb) {
        v16h a  = a_frag(Ht, lane, kb);
        v16h bm = b_frag(w2bT, lane, n0, kb);
        acc2 = __builtin_amdgcn_wmma_f32_16x16x32_f16(false, a, false, bm, (short)0, acc2, false, false);
    }
    // out[b][o][npt]: o = n (this wave's column), npt = p0 + 8g + r (contiguous)
    float* dst = out + ((size_t)b * C_DIM + n) * N_PTS + p0 + 8 * g;
    *(float4*)(dst + 0) = make_float4(acc2[0], acc2[1], acc2[2], acc2[3]);
    *(float4*)(dst + 4) = make_float4(acc2[4], acc2[5], acc2[6], acc2[7]);
}

// ---------------------------------------------------------------------------
extern "C" void kernel_launch(void* const* d_in, const int* in_sizes, int n_in,
                              void* d_out, int out_size, void* d_ws, size_t ws_size,
                              hipStream_t stream) {
    const float* nf  = (const float*)d_in[0];   // [B,128,N]
    const int*   hm  = (const int*)d_in[1];     // [B,K,32]
    const int*   phi = (const int*)d_in[2];     // [B,N,8]
    const float* cen = (const float*)d_in[3];   // [B,128,K]
    const float* w1a = (const float*)d_in[4];
    const float* b1a = (const float*)d_in[5];
    const float* w1b = (const float*)d_in[6];
    const float* b1b = (const float*)d_in[7];
    const float* w2a = (const float*)d_in[8];
    const float* b2a = (const float*)d_in[9];
    const float* w2b = (const float*)d_in[10];
    const float* b2b = (const float*)d_in[11];
    const float* eps = (const float*)d_in[12];
    float* out = (float*)d_out;

    char* ws = (char*)d_ws;
    auto carve = [&](size_t bytes) {
        char* p = ws;
        ws += (bytes + 255) & ~(size_t)255;
        return p;
    };
    _Float16* nfT  = (_Float16*)carve((size_t)B_DIM * N_PTS * C_DIM * 2);   // 25.6 MB
    float*    cenT = (float*)   carve((size_t)B_DIM * K_PAD * C_DIM * 4);   // 5.1 MB
    _Float16* w1aT = (_Float16*)carve((size_t)C_DIM * C_DIM * 2);
    _Float16* w1bT = (_Float16*)carve((size_t)C_DIM * C_DIM * 2);
    _Float16* w2aT = (_Float16*)carve((size_t)C_DIM * C_DIM * 2);
    _Float16* w2bT = (_Float16*)carve((size_t)C_DIM * C_DIM * 2);
    _Float16* heh  = (_Float16*)carve((size_t)B_DIM * K_PAD * C_DIM * 2);   // 2.6 MB

    // 1) weights -> transposed f16
    convert_wT_kernel<<<64, 256, 0, stream>>>(w1a, w1aT);
    convert_wT_kernel<<<64, 256, 0, stream>>>(w1b, w1bT);
    convert_wT_kernel<<<64, 256, 0, stream>>>(w2a, w2aT);
    convert_wT_kernel<<<64, 256, 0, stream>>>(w2b, w2bT);

    // 2) node features [B,C,N] -> f16 [B,N,C]; centers [B,C,K] -> f32 [B,Kp,C] (zero pad)
    transpose_f16_kernel<<<dim3((N_PTS + 31) / 32, C_DIM / 32, B_DIM), dim3(32, 8), 0, stream>>>(
        nf, nfT, N_PTS, N_PTS);
    transpose_f32_kernel<<<dim3((K_PAD + 31) / 32, C_DIM / 32, B_DIM), dim3(32, 8), 0, stream>>>(
        cen, cenT, K_HE, K_PAD);

    // 3) fused gather + hyperedge MLP (WMMA) + centers
    gemm_he_fused<<<B_DIM * (K_PAD / 16), 256, 0, stream>>>(
        nfT, hm, w1aT, b1a, w1bT, b1b, cenT, eps, heh);

    // 4) fused gather + node MLP (WMMA), transposed store to out [B,O,N]
    gemm_out_fused<<<B_DIM * (N_PTS / 16), 256, 0, stream>>>(
        heh, phi, w2aT, b2a, w2bT, b2b, out);
}
